// StochasticLSNN_40956808135047
// MI455X (gfx1250) — matrix operand order repaired and checked by reading, not compiled, not run on hardware
//
#include <hip/hip_runtime.h>
#include <cstdint>

#define T_STEPS 128
#define BATCH   32
#define N_INP   512
#define N_NEU   2048

#define DECAY_F        0.9048374180359595f   // exp(-1/10)
#define ONE_M_DECAY_F  0.09516258196404048f
#define RHO_F          0.9512294245007140f   // exp(-1/20)
#define ONE_M_RHO_F    0.04877057549928599f
#define THR_F          0.03f
#define BETA_F         1.7f
#define REF_F          2.0f
#define EPS_F          1e-6f

// LDS layout for the scan kernel: 32 weight rows padded to 4112B (+4 banks/row
// => conflict-free b128 reads), then the cross-wave reduction buffer.
#define WROW_BYTES     4112
#define WROW_ELEMS     2056
#define SCAN_LDS_BYTES (32 * WROW_BYTES + 4 * 1024)

typedef __attribute__((ext_vector_type(8)))  float  v8f;
typedef __attribute__((ext_vector_type(8)))  __bf16 v8bf;
typedef __attribute__((ext_vector_type(16))) __bf16 v16bf;

#define CAT16(lo, hi) __builtin_shufflevector(lo, hi, 0,1,2,3,4,5,6,7,8,9,10,11,12,13,14,15)

// One 16x16x32 bf16 WMMA step. A-fragment: lane holds row (lane&15),
// K contiguous 16*(lane>=16)+0..15  -> loads at a, a+8.
// B-fragment: lane holds col (lane&15), K contiguous 16*(lane>=16)+0..15
// -> loads at b, b+8. Works for global or LDS-resident B.
static __device__ __forceinline__ v8f wmma_step(const __bf16* a, const __bf16* b, v8f acc) {
  v8bf al = *(const v8bf*)(a);
  v8bf ah = *(const v8bf*)(a + 8);
  v8bf bl = *(const v8bf*)(b);
  v8bf bh = *(const v8bf*)(b + 8);
  v16bf af = CAT16(al, ah);
  v16bf bf = CAT16(bl, bh);
  return __builtin_amdgcn_wmma_f32_16x16x32_bf16(false, af, false, bf, (short)0, acc, false, false);
}

// ---------------------------------------------------------------- init/state
__global__ void k_init(float* st, __bf16* sbuf, unsigned int* bar) {
  size_t i = (size_t)blockIdx.x * blockDim.x + threadIdx.x;
  size_t stride = (size_t)gridDim.x * blockDim.x;
  for (size_t j = i; j < (size_t)4 * BATCH * N_NEU; j += stride) st[j] = 0.0f;
  for (size_t j = i; j < (size_t)2 * BATCH * N_NEU; j += stride) sbuf[j] = (__bf16)0.0f;
  if (i == 0) *bar = 0u;
}

// ------------------------------------------------------------- prep: x->bf16
__global__ void k_prep_x(const float* __restrict__ x, __bf16* __restrict__ xbf) {
  size_t i = (size_t)blockIdx.x * blockDim.x + threadIdx.x;
  if (i < (size_t)T_STEPS * BATCH * N_INP) xbf[i] = (__bf16)x[i];
}

// ---------------------------------------- prep: W_in [512,2048] -> bf16 [n][i]
__global__ void k_prep_win(const float* __restrict__ win, __bf16* __restrict__ wint) {
  size_t idx = (size_t)blockIdx.x * blockDim.x + threadIdx.x;
  if (idx >= (size_t)N_INP * N_NEU) return;
  int i = (int)(idx >> 11);
  int n = (int)(idx & (N_NEU - 1));
  wint[(size_t)n * N_INP + i] = (__bf16)win[idx];   // coalesced read
}

// ------- prep: rwt[n][k] = inh[k] * softplus(k==n ? 0 : Wrec[k][n])  (bf16)
__global__ void k_prep_rw(const float* __restrict__ wrec, const float* __restrict__ inh,
                          __bf16* __restrict__ rwt) {
  size_t tid = (size_t)blockIdx.x * blockDim.x + threadIdx.x;
  if (tid >= (size_t)N_NEU * (N_NEU / 8)) return;
  int n  = (int)(tid & (N_NEU - 1));
  int kb = (int)(tid >> 11);
  v8bf o;
  #pragma unroll
  for (int j = 0; j < 8; ++j) {
    int k = kb * 8 + j;
    float w = (k == n) ? 0.0f : wrec[(size_t)k * N_NEU + n];   // coalesced in n
    float sp = fmaxf(w, 0.0f) + log1pf(expf(-fabsf(w)));       // stable softplus
    o[j] = (__bf16)(inh[k] * sp);
  }
  *(v8bf*)(rwt + (size_t)n * N_NEU + kb * 8) = o;
}

// ------- xw GEMM: [4096,512]bf16 @ [512,2048]bf16 -> f32, + baseline fused
__global__ __launch_bounds__(128)
void k_xw(const __bf16* __restrict__ xbf, const __bf16* __restrict__ wint,
          const float* __restrict__ baseline, float* __restrict__ xw) {
  const int lane = threadIdx.x & 31;
  const int wave = threadIdx.x >> 5;
  const int id   = (blockIdx.x << 2) | wave;   // 0..32767 tiles
  const int mt   = id >> 7;
  const int nt   = id & 127;
  const int fr   = lane & 15;
  const int hi   = lane >> 4;
  const __bf16* arow = xbf  + (size_t)((mt << 4) + fr) * N_INP + (hi << 4);
  const __bf16* brow = wint + (size_t)((nt << 4) + fr) * N_INP + (hi << 4);
  v8f acc0 = {0,0,0,0,0,0,0,0};
  v8f acc1 = {0,0,0,0,0,0,0,0};
  #pragma unroll 2
  for (int kk = 0; kk < N_INP; kk += 64) {
    acc0 = wmma_step(arow + kk,      brow + kk,      acc0);
    acc1 = wmma_step(arow + kk + 32, brow + kk + 32, acc1);
  }
  v8f acc = acc0 + acc1;
  const int nn = (nt << 4) + fr;
  const int mb = (mt << 4) + (hi << 3);
  const float bl = baseline[nn];
  #pragma unroll
  for (int r = 0; r < 8; ++r)
    xw[(size_t)(mb + r) * N_NEU + nn] = acc[r] + bl;
}

// --------------------------------------------------------- persistent scan
__global__ __launch_bounds__(256)
void k_scan(const __bf16* __restrict__ rwt, const float* __restrict__ xw,
            const float* __restrict__ noise, const float* __restrict__ logtemp,
            const int* __restrict__ bzzp, float* __restrict__ out,
            __bf16* __restrict__ sbuf, float* __restrict__ st,
            unsigned int* __restrict__ bar) {
  extern __shared__ char smem[];
  __bf16* wlds = (__bf16*)smem;                    // 32 rows, padded stride
  float*  red  = (float*)(smem + 32 * WROW_BYTES); // 4 tiles * 256 floats

  const int tid   = threadIdx.x;
  const int lane  = tid & 31;
  const int wave  = tid >> 5;
  const int tile  = wave & 3;      // which 16x16 tile within WG
  const int khalf = wave >> 2;     // K split across wave pairs
  const int mtile = tile & 1;
  const int nloc  = (tile >> 1) << 4;               // local col base (0/16)
  const int m0 = mtile << 4;
  const int fr = lane & 15;
  const int hi = lane >> 4;
  const int kA = hi << 4;          // A frag K base (0/16)
  const int kB = hi << 4;          // B frag K base (0/16)
  const int k0 = khalf << 10;

  float* vst  = st;
  float* ast  = st + 1 * BATCH * N_NEU;
  float* lsds = st + 2 * BATCH * N_NEU;
  float* spks = st + 3 * BATCH * N_NEU;

  // ---- one-time async fill: this WG's 32 rwt rows -> LDS (ASYNCcnt path)
  {
    const int gbase = blockIdx.x * 32;              // global rwt row base
    const unsigned ldsb = (unsigned)(uintptr_t)wlds;
    #pragma unroll 4
    for (int r = 0; r < 32; ++r) {
      uint64_t ga = (uint64_t)(uintptr_t)(rwt + (size_t)(gbase + r) * N_NEU + tid * 8);
      unsigned  la = ldsb + (unsigned)(r * WROW_BYTES + tid * 16);
      asm volatile("global_load_async_to_lds_b128 %0, %1, off"
                   :: "v"(la), "v"(ga) : "memory");
    }
    asm volatile("s_wait_asynccnt 0x0" ::: "memory");
  }
  __syncthreads();

  const float inv_temp = expf(-logtemp[0]);
  const int bzz = *bzzp;
  const unsigned int nwg = gridDim.x;
  const int nn = (blockIdx.x * 32) + nloc + fr;     // global output column
  const int mb = m0 + (hi << 3);
  const float beta = (nn < (N_NEU / 2)) ? 0.0f : BETA_F;
  const __bf16* brow = wlds + (size_t)(nloc + fr) * WROW_ELEMS + k0 + kB;

  for (int t = 0; t < T_STEPS; ++t) {
    const __bf16* sin  = sbuf + (size_t)(t & 1) * (BATCH * N_NEU);
    __bf16*       sout = sbuf + (size_t)((t + 1) & 1) * (BATCH * N_NEU);
    const __bf16* arow = sin + (size_t)(m0 + fr) * N_NEU + k0 + kA;
    const size_t gb = ((size_t)t * BATCH + mb) * N_NEU + nn;

    // prefetch this step's HBM operands; latency hides behind the GEMM
    float uu[8], xv[8];
    if (khalf == 0) {
      #pragma unroll
      for (int r = 0; r < 8; ++r) {
        uu[r] = noise[gb + (size_t)r * N_NEU];
        xv[r] = xw[gb + (size_t)r * N_NEU];
      }
    }

    v8f acc0 = {0,0,0,0,0,0,0,0};
    v8f acc1 = {0,0,0,0,0,0,0,0};
    #pragma unroll 2
    for (int kk = 0; kk < 1024; kk += 64) {
      acc0 = wmma_step(arow + kk,      brow + kk,      acc0);   // B from LDS
      acc1 = wmma_step(arow + kk + 32, brow + kk + 32, acc1);
    }
    v8f acc = acc0 + acc1;

    // cross-wave K reduction through LDS
    if (khalf == 1) {
      #pragma unroll
      for (int r = 0; r < 8; ++r) red[tile * 256 + r * 32 + lane] = acc[r];
    }
    __syncthreads();

    if (khalf == 0) {
      #pragma unroll
      for (int r = 0; r < 8; ++r) {
        const int b = mb + r;
        const size_t sidx = (size_t)b * N_NEU + nn;
        const size_t gidx = gb + (size_t)r * N_NEU;
        float rec   = acc[r] + red[tile * 256 + r * 32 + lane];
        float spike = spks[sidx];
        float vv    = vst[sidx];
        float aa    = ast[sidx];
        float lsd   = lsds[sidx];
        float u     = uu[r];
        float i_in  = xv[r] + rec;                    // baseline folded into xw
        float new_a = RHO_F * aa + ONE_M_RHO_F * spike;
        float thr   = THR_F + new_a * beta;
        float new_v = DECAY_F * vv + ONE_M_DECAY_F * i_in - thr * spike;
        float la    = (new_v - thr) / thr;            // log_alpha
        float z     = la + logf(u) - log1pf(-u);
        float hard0 = (z > 0.0f) ? 1.0f : 0.0f;
        float hard  = (lsd > REF_F) ? hard0 : 0.0f;   // refractory mask
        float o;
        if (bzz == 0) {
          o = hard;
        } else if (bzz == 1) {
          o = 1.0f / (1.0f + expf(-z * inv_temp));
        } else {
          float up = 1.0f / (1.0f + expf(la));        // sigmoid(-la)
          float vc = (hard0 > 0.0f) ? (up + u * (1.0f - up)) : (u * up);
          vc = fminf(fmaxf(vc, EPS_F), 1.0f - EPS_F);
          float zt = la + logf(vc) - log1pf(-vc);
          o = 1.0f / (1.0f + expf(-zt * inv_temp));
        }
        out[gidx]  = o;
        spks[sidx] = o;
        vst[sidx]  = new_v;
        ast[sidx]  = new_a;
        lsds[sidx] = (lsd + 1.0f) * (1.0f - hard);
        sout[sidx] = (__bf16)o;                       // next step's A matrix
      }
    }

    // grid-wide barrier: all spike writes visible before next step's GEMM
    __syncthreads();
    if (tid == 0) {
      __threadfence();
      __hip_atomic_fetch_add(bar, 1u, __ATOMIC_ACQ_REL, __HIP_MEMORY_SCOPE_AGENT);
      const unsigned int target = (unsigned int)(t + 1) * nwg;
      while (__hip_atomic_load(bar, __ATOMIC_ACQUIRE, __HIP_MEMORY_SCOPE_AGENT) < target) {
        __builtin_amdgcn_s_sleep(2);
      }
    }
    __syncthreads();
  }
}

// ------------------------------------------------------------------ launch
extern "C" void kernel_launch(void* const* d_in, const int* in_sizes, int n_in,
                              void* d_out, int out_size, void* d_ws, size_t ws_size,
                              hipStream_t stream) {
  (void)in_sizes; (void)n_in; (void)out_size; (void)ws_size;
  const float* x        = (const float*)d_in[0];
  const float* noise    = (const float*)d_in[1];
  const float* win      = (const float*)d_in[2];
  const float* wrec     = (const float*)d_in[3];
  const float* baseline = (const float*)d_in[4];
  const float* inh      = (const float*)d_in[5];
  const float* logtemp  = (const float*)d_in[6];
  const int*   bzz      = (const int*)d_in[7];

  char* ws = (char*)d_ws;
  size_t o = 0;
  __bf16* rwt  = (__bf16*)(ws + o); o += (size_t)N_NEU * N_NEU * 2;            //  8 MB
  __bf16* wint = (__bf16*)(ws + o); o += (size_t)N_NEU * N_INP * 2;            //  2 MB
  __bf16* xbf  = (__bf16*)(ws + o); o += (size_t)T_STEPS * BATCH * N_INP * 2;  //  4 MB
  float*  xw   = (float*)(ws + o);  o += (size_t)T_STEPS * BATCH * N_NEU * 4;  // 32 MB
  __bf16* sbuf = (__bf16*)(ws + o); o += (size_t)2 * BATCH * N_NEU * 2;        // ping-pong spikes
  float*  st   = (float*)(ws + o);  o += (size_t)4 * BATCH * N_NEU * 4;        // v,a,lsd,spike
  unsigned int* bar = (unsigned int*)(ws + o);

  k_init<<<256, 256, 0, stream>>>(st, sbuf, bar);
  k_prep_x<<<(T_STEPS * BATCH * N_INP + 255) / 256, 256, 0, stream>>>(x, xbf);
  k_prep_win<<<(N_INP * N_NEU + 255) / 256, 256, 0, stream>>>(win, wint);
  k_prep_rw<<<(N_NEU * (N_NEU / 8) + 255) / 256, 256, 0, stream>>>(wrec, inh, rwt);
  k_xw<<<8192, 128, 0, stream>>>(xbf, wint, baseline, xw);
  k_scan<<<64, 256, SCAN_LDS_BYTES, stream>>>(rwt, xw, noise, logtemp, bzz,
                                              (float*)d_out, sbuf, st, bar);
}
